// Encoder_16578573763343
// MI455X (gfx1250) — matrix operand order repaired and checked by reading, compiled, not run on hardware
//
#include <hip/hip_runtime.h>
#include <hip/hip_bf16.h>

#define S_LEN   2048
#define D_MODEL 1024
#define N_HEADS 8
#define D_FF    3072

typedef __attribute__((ext_vector_type(16))) __bf16 v16bf;
typedef __attribute__((ext_vector_type(8)))  float  v8f;
typedef __attribute__((ext_vector_type(4)))  float  f32x4;
typedef __attribute__((ext_vector_type(2)))  float  f32x2;
typedef __attribute__((ext_vector_type(2)))  unsigned u32x2;
typedef __attribute__((ext_vector_type(4)))  unsigned u32x4;

// bf16 round-to-nearest-even, all 32-bit ALU
__device__ __forceinline__ unsigned bf16_rne(float f) {
  union { float f; unsigned u; } v; v.f = f;
  return v.u + 0x7FFFu + ((v.u >> 16) & 1u);   // caller takes bits [31:16]
}
// pack two f32 -> (bf16(b)<<16 | bf16(a))
__device__ __forceinline__ unsigned pack_bf16x2(float a, float b) {
  return (bf16_rne(a) >> 16) | (bf16_rne(b) & 0xFFFF0000u);
}

enum { EPI_BF16_BIAS = 0, EPI_F32_SCALE = 1, EPI_F32_ACC = 2,
       EPI_BF16_RELU_BIAS = 3, EPI_F32_BIAS = 4 };

union FragU { unsigned u[8]; v16bf v; };

// ---------------------------------------------------------------------------
// Generic bf16-WMMA GEMM:  O[M,N] = epi( A[M,K] * B + bias )
//   A row-major [M][K]  (f32 or bf16 source, packed to bf16 in LDS)
//   B_TRANS=0: B is [N][K] row-major (Q@K^T uses K-matrix rows directly)
//   B_TRANS=1: B is [K][N] row-major (weights, V) -> transposed while staging
// Block tile 128x128, 8 waves (4x2), wave tile 32x64 = 2x4 wmma, K step 32.
// All index math 32-bit (every operand < 2^31 elements).
// ---------------------------------------------------------------------------
template<int A_BF16, int B_BF16, int B_TRANS, int EPI>
__global__ __launch_bounds__(256) void gemm_wmma(
    const void* __restrict__ Ag, const void* __restrict__ Bg,
    const float* __restrict__ bias, void* __restrict__ Og,
    int M, int N, int K, float alpha, int accumulate)
{
  __shared__ unsigned short As[128][36];   // [m][k] bf16, 72B rows (8B aligned)
  __shared__ unsigned short Bs[128][36];   // [n][k] bf16

  const int tid   = threadIdx.x;
  const int lane  = tid & 31;
  const int wave  = tid >> 5;
  const int waveM = wave >> 1;   // 0..3 -> 32-row slab
  const int waveN = wave & 1;    // 0..1 -> 64-col slab
  const int l15   = lane & 15;
  const int lhi   = lane >> 4;   // 0 or 1

  const int blockM = blockIdx.y * 128;
  const int blockN = blockIdx.x * 128;

  // per-thread staging decomposition (constant across k-chunks)
  const int ar  = tid >> 3;          // A/Bdirect base row (0..31), rows r+32*i
  const int ac  = (tid & 7) * 4;     // A/Bdirect col (0,4,...,28)
  const int tkb = (tid >> 6) * 2;    // Btrans base k (0,2,4,6), k = tkb+8*i
  const int tn  = (tid & 63) * 2;    // Btrans col pair (0..126)

  v8f acc[2][4] = {};

  for (int k0 = 0; k0 < K; k0 += 32) {
    // ---- prefetch next chunk (global_prefetch_b8) ----
    if (k0 + 32 < K) {
      int anext = (blockM + ar) * K + k0 + 32 + ac;
      __builtin_prefetch((const char*)Ag + (size_t)anext * (A_BF16 ? 2 : 4), 0, 1);
      int bnext = B_TRANS ? ((k0 + 32 + tkb) * N + blockN + tn)
                          : ((blockN + ar) * K + k0 + 32 + ac);
      __builtin_prefetch((const char*)Bg + (size_t)bnext * (B_BF16 ? 2 : 4), 0, 1);
    }

    // ---- stage A tile: 128 x 32, 4 elems/thread/iter, b64 LDS stores ----
#pragma unroll
    for (int i = 0; i < 4; ++i) {
      int r = ar + 32 * i;
      int gidx = (blockM + r) * K + k0 + ac;
      u32x2 p;
      if (A_BF16) {
        p = *(const u32x2*)((const unsigned short*)Ag + gidx);
      } else {
        f32x4 d = *(const f32x4*)((const float*)Ag + gidx);
        p.x = pack_bf16x2(d.x, d.y);
        p.y = pack_bf16x2(d.z, d.w);
      }
      *(u32x2*)&As[r][ac] = p;
    }

    // ---- stage B tile into [n][k] ----
    if (B_TRANS) {
      // B is [K][N]: read rows k,k+1 at cols n,n+1; pack k-pairs -> b32 stores
#pragma unroll
      for (int i = 0; i < 4; ++i) {
        int kk = tkb + 8 * i;
        int gidx = (k0 + kk) * N + blockN + tn;
        if (B_BF16) {
          unsigned d0 = *(const unsigned*)((const unsigned short*)Bg + gidx);
          unsigned d1 = *(const unsigned*)((const unsigned short*)Bg + gidx + N);
          *(unsigned*)&Bs[tn + 0][kk] = (d0 & 0xFFFFu) | (d1 << 16);
          *(unsigned*)&Bs[tn + 1][kk] = (d0 >> 16) | (d1 & 0xFFFF0000u);
        } else {
          f32x2 r0 = *(const f32x2*)((const float*)Bg + gidx);
          f32x2 r1 = *(const f32x2*)((const float*)Bg + gidx + N);
          *(unsigned*)&Bs[tn + 0][kk] = pack_bf16x2(r0.x, r1.x);
          *(unsigned*)&Bs[tn + 1][kk] = pack_bf16x2(r0.y, r1.y);
        }
      }
    } else {
      // B is [N][K]: same shape as A tile
#pragma unroll
      for (int i = 0; i < 4; ++i) {
        int n = ar + 32 * i;
        int gidx = (blockN + n) * K + k0 + ac;
        u32x2 p;
        if (B_BF16) {
          p = *(const u32x2*)((const unsigned short*)Bg + gidx);
        } else {
          f32x4 d = *(const f32x4*)((const float*)Bg + gidx);
          p.x = pack_bf16x2(d.x, d.y);
          p.y = pack_bf16x2(d.z, d.w);
        }
        *(u32x2*)&Bs[n][ac] = p;
      }
    }
    __syncthreads();

    // ---- build fragments per ISA 7.12.2 layouts ----
    FragU a[2], b[4];
#pragma unroll
    for (int mi = 0; mi < 2; ++mi) {
      int m = waveM * 32 + mi * 16 + l15;
#pragma unroll
      for (int j = 0; j < 8; ++j) {
        int k = 16 * (j >> 2) + 8 * lhi + 2 * (j & 3); // A: K 0..7 lanes0-15, 8..15 lanes16-31, +16
        a[mi].u[j] = *(const unsigned*)&As[m][k];
      }
    }
#pragma unroll
    for (int ni = 0; ni < 4; ++ni) {
      int n = waveN * 64 + ni * 16 + l15;
#pragma unroll
      for (int j = 0; j < 8; ++j) {
        int k = 16 * (j >> 2) + 8 * lhi + 2 * (j & 3); // B: K 0..7 lanes0-15, 8..15 lanes16-31, +16
        b[ni].u[j] = *(const unsigned*)&Bs[n][k];
      }
    }

#pragma unroll
    for (int mi = 0; mi < 2; ++mi)
#pragma unroll
      for (int ni = 0; ni < 4; ++ni)
        acc[mi][ni] = __builtin_amdgcn_wmma_f32_16x16x32_bf16(
            false, a[mi].v, false, b[ni].v, (short)0, acc[mi][ni], false, false);

    __syncthreads();
  }

  // ---- epilogue: C/D layout  m = r + 8*(lane>>4), n = lane&15 ----
  const int obase = (blockM + waveM * 32 + 8 * lhi) * N + blockN + waveN * 64 + l15;
#pragma unroll
  for (int mi = 0; mi < 2; ++mi) {
#pragma unroll
    for (int ni = 0; ni < 4; ++ni) {
      int col = blockN + waveN * 64 + ni * 16 + l15;
#pragma unroll
      for (int r = 0; r < 8; ++r) {
        int oidx = obase + (mi * 16 + r) * N + ni * 16;
        float val = acc[mi][ni][r];
        if (EPI == EPI_F32_SCALE) {
          ((float*)Og)[oidx] = val * alpha;
        } else if (EPI == EPI_BF16_BIAS) {
          ((unsigned short*)Og)[oidx] =
              (unsigned short)(bf16_rne(val + bias[col]) >> 16);
        } else if (EPI == EPI_BF16_RELU_BIAS) {
          float t = val + bias[col];
          ((unsigned short*)Og)[oidx] =
              (unsigned short)(bf16_rne(t > 0.0f ? t : 0.0f) >> 16);
        } else if (EPI == EPI_F32_BIAS) {
          ((float*)Og)[oidx] = val + bias[col];
        } else { // EPI_F32_ACC
          float* o = (float*)Og;
          o[oidx] = accumulate ? (o[oidx] + val) : val;
        }
      }
    }
  }
}

// ---------------------------------------------------------------------------
// x[s][d] = emb[tokens[s]][d] + PE(s,d)
// ---------------------------------------------------------------------------
__global__ __launch_bounds__(256) void embed_pe(
    const int* __restrict__ tokens, const float* __restrict__ emb,
    float* __restrict__ x)
{
  int s = blockIdx.y;
  int d = blockIdx.x * 256 + threadIdx.x;
  float e = emb[(size_t)tokens[s] * D_MODEL + d];
  int i2 = d & ~1;
  float freq  = __expf(-(float)i2 * (9.2103403719761836f / (float)D_MODEL));
  float angle = (float)s * freq;
  float pe = (d & 1) ? __cosf(angle) : __sinf(angle);
  x[s * D_MODEL + d] = e + pe;
}

// ---------------------------------------------------------------------------
// row softmax over 2048 f32 -> bf16 probs (contiguous 8-elem strips/thread)
// ---------------------------------------------------------------------------
__global__ __launch_bounds__(256) void softmax_row(
    const float* __restrict__ sc, unsigned short* __restrict__ attn)
{
  __shared__ float red[256];
  const int tid = threadIdx.x;
  const int base = blockIdx.x * S_LEN + tid * 8;

  f32x4 d0 = *(const f32x4*)(sc + base);
  f32x4 d1 = *(const f32x4*)(sc + base + 4);
  float m = fmaxf(fmaxf(fmaxf(d0.x, d0.y), fmaxf(d0.z, d0.w)),
                  fmaxf(fmaxf(d1.x, d1.y), fmaxf(d1.z, d1.w)));
  red[tid] = m; __syncthreads();
  for (int s = 128; s > 0; s >>= 1) {
    if (tid < s) red[tid] = fmaxf(red[tid], red[tid + s]);
    __syncthreads();
  }
  m = red[0]; __syncthreads();

  float e[8];
  e[0]=__expf(d0.x-m); e[1]=__expf(d0.y-m); e[2]=__expf(d0.z-m); e[3]=__expf(d0.w-m);
  e[4]=__expf(d1.x-m); e[5]=__expf(d1.y-m); e[6]=__expf(d1.z-m); e[7]=__expf(d1.w-m);
  float sum = ((e[0]+e[1])+(e[2]+e[3]))+((e[4]+e[5])+(e[6]+e[7]));
  red[tid] = sum; __syncthreads();
  for (int s = 128; s > 0; s >>= 1) {
    if (tid < s) red[tid] += red[tid + s];
    __syncthreads();
  }
  float inv = 1.0f / red[0];
  u32x4 p;
  p.x = pack_bf16x2(e[0]*inv, e[1]*inv);
  p.y = pack_bf16x2(e[2]*inv, e[3]*inv);
  p.z = pack_bf16x2(e[4]*inv, e[5]*inv);
  p.w = pack_bf16x2(e[6]*inv, e[7]*inv);
  *(u32x4*)(attn + base) = p;
}

// ---------------------------------------------------------------------------
// z = mha + x (in place into z buffer), per-block partial sum/sumsq
// ---------------------------------------------------------------------------
__global__ __launch_bounds__(256) void resid_stats(
    const float* __restrict__ x, float* __restrict__ z,
    float* __restrict__ partials)
{
  __shared__ float r1[256], r2[256];
  const int tid = threadIdx.x;
  const int base = blockIdx.x * 2048 + tid * 8;
  float s = 0.0f, s2 = 0.0f;
#pragma unroll
  for (int i = 0; i < 2; ++i) {
    f32x4 a = *(const f32x4*)(z + base + 4 * i);
    f32x4 b = *(const f32x4*)(x + base + 4 * i);
    f32x4 v = a + b;
    *(f32x4*)(z + base + 4 * i) = v;
    s  += ((v.x + v.y) + (v.z + v.w));
    s2 += ((v.x*v.x + v.y*v.y) + (v.z*v.z + v.w*v.w));
  }
  r1[tid] = s; r2[tid] = s2; __syncthreads();
  for (int k = 128; k > 0; k >>= 1) {
    if (tid < k) { r1[tid] += r1[tid + k]; r2[tid] += r2[tid + k]; }
    __syncthreads();
  }
  if (tid == 0) { partials[2 * blockIdx.x] = r1[0]; partials[2 * blockIdx.x + 1] = r2[0]; }
}

__global__ __launch_bounds__(256) void finalize_stats(
    const float* __restrict__ partials, float* __restrict__ stats)
{
  __shared__ float r1[256], r2[256];
  const int tid = threadIdx.x;
  float s = 0.0f, s2 = 0.0f;
  for (int i = tid; i < 1024; i += 256) { s += partials[2 * i]; s2 += partials[2 * i + 1]; }
  r1[tid] = s; r2[tid] = s2; __syncthreads();
  for (int k = 128; k > 0; k >>= 1) {
    if (tid < k) { r1[tid] += r1[tid + k]; r2[tid] += r2[tid + k]; }
    __syncthreads();
  }
  if (tid == 0) {
    const float n = (float)S_LEN * (float)D_MODEL;
    float mean = r1[0] / n;
    float var  = r2[0] / n - mean * mean;
    stats[0] = mean;
    stats[1] = rsqrtf(var + 1e-5f);
  }
}

__global__ __launch_bounds__(256) void normalize_k(
    const float* __restrict__ z, const float* __restrict__ stats,
    unsigned short* __restrict__ zn)
{
  int i = (blockIdx.x * 256 + threadIdx.x) * 4;
  float mean = stats[0], rstd = stats[1];
  f32x4 v = *(const f32x4*)(z + i);
  u32x2 p;
  p.x = pack_bf16x2((v.x - mean) * rstd, (v.y - mean) * rstd);
  p.y = pack_bf16x2((v.z - mean) * rstd, (v.w - mean) * rstd);
  *(u32x2*)(zn + i) = p;
}

// ---------------------------------------------------------------------------
extern "C" void kernel_launch(void* const* d_in, const int* in_sizes, int n_in,
                              void* d_out, int out_size, void* d_ws, size_t ws_size,
                              hipStream_t stream) {
  (void)in_sizes; (void)n_in; (void)out_size; (void)ws_size;

  const int*   tokens = (const int*)  d_in[0];
  const float* emb    = (const float*)d_in[1];
  const float* Wq     = (const float*)d_in[2];
  const float* bq     = (const float*)d_in[3];
  const float* Wk     = (const float*)d_in[4];
  const float* bk     = (const float*)d_in[5];
  const float* Wv     = (const float*)d_in[6];
  const float* bv     = (const float*)d_in[7];
  const float* W1     = (const float*)d_in[8];
  const float* b1     = (const float*)d_in[9];
  const float* W2     = (const float*)d_in[10];
  const float* b2     = (const float*)d_in[11];

  char* ws = (char*)d_ws;
  size_t off = 0;
  auto carve = [&](size_t bytes) -> void* {
    void* p = ws + off;
    off += (bytes + 255) & ~(size_t)255;
    return p;
  };
  float*          x       = (float*)         carve((size_t)S_LEN * D_MODEL * 4);
  unsigned short* qb      = (unsigned short*)carve((size_t)N_HEADS * S_LEN * D_MODEL * 2);
  unsigned short* kb      = (unsigned short*)carve((size_t)N_HEADS * S_LEN * D_MODEL * 2);
  unsigned short* vb      = (unsigned short*)carve((size_t)N_HEADS * S_LEN * D_MODEL * 2);
  float*          scores  = (float*)         carve((size_t)S_LEN * S_LEN * 4);
  unsigned short* attn    = (unsigned short*)carve((size_t)S_LEN * S_LEN * 2);
  float*          mha     = (float*)         carve((size_t)S_LEN * D_MODEL * 4);
  unsigned short* zn      = (unsigned short*)carve((size_t)S_LEN * D_MODEL * 2);
  unsigned short* hb      = (unsigned short*)carve((size_t)S_LEN * D_FF * 2);
  float*          partials= (float*)         carve(1024 * 2 * 4);
  float*          stats   = (float*)         carve(256);

  dim3 blk(256);

  // x = embed + positional encoding
  embed_pe<<<dim3(D_MODEL / 256, S_LEN), blk, 0, stream>>>(tokens, emb, x);

  // Q/K/V projections: per head  [S,D] = x @ W[h] + b[h]  (bf16 out)
  for (int m = 0; m < 3; ++m) {
    const float* W = (m == 0) ? Wq : (m == 1) ? Wk : Wv;
    const float* b = (m == 0) ? bq : (m == 1) ? bk : bv;
    unsigned short* out = (m == 0) ? qb : (m == 1) ? kb : vb;
    for (int h = 0; h < N_HEADS; ++h) {
      gemm_wmma<0, 0, 1, EPI_BF16_BIAS>
          <<<dim3(D_MODEL / 128, S_LEN / 128), blk, 0, stream>>>(
          x, W + (size_t)h * D_MODEL * D_MODEL, b + (size_t)h * D_MODEL,
          out + (size_t)h * S_LEN * D_MODEL, S_LEN, D_MODEL, D_MODEL, 1.0f, 0);
    }
  }

  // Attention, head by head (score/attn buffers reused)
  for (int h = 0; h < N_HEADS; ++h) {
    const unsigned short* qh = qb + (size_t)h * S_LEN * D_MODEL;
    const unsigned short* kh = kb + (size_t)h * S_LEN * D_MODEL;
    const unsigned short* vh = vb + (size_t)h * S_LEN * D_MODEL;
    // scores = (q @ k^T) / 32    (B direct: k rows are already [n][k])
    gemm_wmma<1, 1, 0, EPI_F32_SCALE>
        <<<dim3(S_LEN / 128, S_LEN / 128), blk, 0, stream>>>(
        qh, kh, nullptr, scores, S_LEN, S_LEN, D_MODEL, 0.03125f, 0);
    softmax_row<<<dim3(S_LEN), blk, 0, stream>>>(scores, attn);
    // mha (+)= attn @ v          (B trans: v is [K=t][N=e])
    gemm_wmma<1, 1, 1, EPI_F32_ACC>
        <<<dim3(D_MODEL / 128, S_LEN / 128), blk, 0, stream>>>(
        attn, vh, nullptr, mha, S_LEN, D_MODEL, S_LEN, 1.0f, h > 0 ? 1 : 0);
  }

  // z = mha + x; whole-tensor layernorm -> zn (bf16)
  resid_stats<<<dim3(1024), blk, 0, stream>>>(x, mha, partials);
  finalize_stats<<<dim3(1), blk, 0, stream>>>(partials, stats);
  normalize_k<<<dim3((S_LEN * D_MODEL) / 1024), blk, 0, stream>>>(mha, stats, zn);

  // FFN
  gemm_wmma<1, 0, 1, EPI_BF16_RELU_BIAS>
      <<<dim3(D_FF / 128, S_LEN / 128), blk, 0, stream>>>(
      zn, W1, b1, hb, S_LEN, D_FF, D_MODEL, 1.0f, 0);
  gemm_wmma<1, 0, 1, EPI_F32_BIAS>
      <<<dim3(D_MODEL / 128, S_LEN / 128), blk, 0, stream>>>(
      hb, W2, b2, (float*)d_out, S_LEN, D_MODEL, D_FF, 1.0f, 0);
}
